// BidirRNNLayer_81647328297082
// MI455X (gfx1250) — matrix operand order repaired and checked by reading, not compiled, not run on hardware
//
#include <hip/hip_runtime.h>
#include <hip/hip_bf16.h>
#include <math.h>

typedef __attribute__((ext_vector_type(16))) __bf16 v16bf;
typedef __attribute__((ext_vector_type(8)))  __bf16 v8bf;
typedef __attribute__((ext_vector_type(4)))  __bf16 v4bf;
typedef __attribute__((ext_vector_type(8)))  float  v8f;

#define RNN_B 64
#define RNN_T 2048
#define RNN_D 256
#define RNN_H 256
#define HPAD  264   // 256 + 8 bf16 (16B) row pad -> LDS bank-conflict-free b128 frag loads

// Hardware tanh (V_TANH_F32) if the toolchain exposes it; else ocml fallback.
#if __has_builtin(__builtin_amdgcn_tanhf)
__device__ __forceinline__ float fast_tanh(float x) { return __builtin_amdgcn_tanhf(x); }
#elif __has_builtin(__builtin_amdgcn_tanh_f32)
__device__ __forceinline__ float fast_tanh(float x) { return __builtin_amdgcn_tanh_f32(x); }
#else
__device__ __forceinline__ float fast_tanh(float x) { return tanhf(x); }
#endif

// ---- WMMA fragment loaders (bf16, 16x16x32), layouts per CDNA5 ISA 7.12.2 ----
// A (16x32, row-major source, ld in elements): lanes 0-15 row M=lane, K {0..7,16..23};
// lanes 16-31 same rows, K {8..15,24..31}. Two 16-byte loads per lane.
__device__ __forceinline__ v16bf frag_a(const __bf16* base, int ld, int lane) {
  const int r = lane & 15, hf = lane >> 4;
  const __bf16* p = base + r * ld + hf * 8;
  v8bf lo = *(const v8bf*)(p);
  v8bf hi = *(const v8bf*)(p + 16);
  return __builtin_shufflevector(lo, hi, 0,1,2,3,4,5,6,7,8,9,10,11,12,13,14,15);
}
// B (32x16): column n = lane&15; lanes 0-15 K {0..7,16..23}, lanes 16-31 K {8..15,24..31}.
// Since out = act * W^T, B column n == W row n (contiguous in memory). base = W + n0*ld + k0.
__device__ __forceinline__ v16bf frag_b(const __bf16* base, int ld, int lane) {
  const int r = lane & 15, hf = lane >> 4;
  const __bf16* p = base + r * ld + hf * 8;
  v8bf lo = *(const v8bf*)(p);
  v8bf hi = *(const v8bf*)(p + 16);
  return __builtin_shufflevector(lo, hi, 0,1,2,3,4,5,6,7,8,9,10,11,12,13,14,15);
}

// ---- Kernel 0a: convert both W_ih matrices to bf16 (2 * 256*256 elements) ----
__global__ void k_conv_wih(const float* __restrict__ wf, const float* __restrict__ wr,
                           __bf16* __restrict__ dst) {
  int i = blockIdx.x * blockDim.x + threadIdx.x;       // exactly 131072 threads
  const int n = RNN_H * RNN_D;
  dst[i] = (__bf16)((i < n) ? wf[i] : wr[i - n]);
}

// ---- Kernel 0b: x [B][T][D] f32 -> xbf [T][B][D] bf16 (coalesced both sides) ----
__global__ void k_conv_x(const float* __restrict__ x, __bf16* __restrict__ xbf) {
  long long i = (long long)blockIdx.x * blockDim.x + threadIdx.x;   // 8,388,608 threads
  long long e = i * 4;
  int d = (int)(e & (RNN_D - 1));
  long long bt = e >> 8;                 // b*T + t
  int t = (int)(bt & (RNN_T - 1));
  int b = (int)(bt >> 11);
  const float4 v = *(const float4*)(x + e);
  v4bf o4 = { (__bf16)v.x, (__bf16)v.y, (__bf16)v.z, (__bf16)v.w };
  *(v4bf*)(xbf + (((long long)t * RNN_B + b) * RNN_D + d)) = o4;
}

// ---- Kernel 1: x_proj GEMM. Block = 4 waves; block owns 16 rows x 256 cols.
// Output stored in C-fragment order: xp[dir][t][mt][nt][lane][r] so the
// recurrent kernel consumes it with one contiguous 32B load per lane per tile.
__global__ __launch_bounds__(128)
void k_xproj(const __bf16* __restrict__ xbf, const __bf16* __restrict__ wih2,
             const float* __restrict__ b_f, const float* __restrict__ b_r,
             float* __restrict__ xp) {
  const int dir  = blockIdx.y;
  const int m0   = blockIdx.x * 16;      // row tile start in flat [T*B] (t-major)
  const int t    = m0 >> 6;
  const int mt   = (m0 >> 4) & 3;
  const int wave = threadIdx.x >> 5;
  const int lane = threadIdx.x & 31;
  const __bf16* wih  = wih2 + (size_t)dir * (RNN_H * RNN_D);
  const float*  bias = dir ? b_r : b_f;
  const __bf16* arow = xbf + (size_t)m0 * RNN_D;

  v8f acc[4] = {};
  #pragma unroll
  for (int k = 0; k < RNN_D; k += 32) {
    v16bf a = frag_a(arow + k, RNN_D, lane);
    #pragma unroll
    for (int j = 0; j < 4; ++j) {
      const int n0 = (wave * 4 + j) * 16;
      v16bf b = frag_b(wih + (size_t)n0 * RNN_D + k, RNN_D, lane);
      acc[j] = __builtin_amdgcn_wmma_f32_16x16x32_bf16(false, a, false, b,
                                                       (short)0, acc[j], false, false);
    }
  }
  float* xpd = xp + (size_t)dir * ((size_t)RNN_T * RNN_B * RNN_H);
  #pragma unroll
  for (int j = 0; j < 4; ++j) {
    const int nt = wave * 4 + j;
    const float bn = bias[nt * 16 + (lane & 15)];
    v8f v = acc[j];
    #pragma unroll
    for (int r = 0; r < 8; ++r) v[r] += bn;
    *(v8f*)(xpd + ((((size_t)t * 4 + mt) * 16 + nt) * 32 + lane) * 8) = v;
  }
}

// ---- Kernel 2: persistent recurrent scan. One block (16 waves) per direction.
// W_hh lives in LDS (bf16), h double-buffered in LDS, one barrier per step.
// xp addend loads are issued BEFORE the WMMA k-loop so their latency is hidden
// behind the matrix work of the same step.
__global__ __launch_bounds__(512, 1)
void k_rnn(const float* __restrict__ hid0,
           const float* __restrict__ whh_f, const float* __restrict__ whh_r,
           const float* __restrict__ xp, float* __restrict__ out) {
  const int dir = blockIdx.x;
  extern __shared__ __bf16 smem[];
  __bf16* Wl  = smem;                  // [256][HPAD]
  __bf16* hb0 = Wl + 256 * HPAD;       // [64][HPAD]
  __bf16* hb1 = hb0 + 64 * HPAD;       // [64][HPAD]

  const int tid  = threadIdx.x;
  const int lane = tid & 31;
  const int wave = tid >> 5;
  const int mt   = wave & 3;           // output m-tile (16 batch rows)
  const int ntb  = (wave >> 2) * 4;    // first of 4 n-tiles
  const int hf   = lane >> 4;
  const int nl   = lane & 15;

  const float* whh = dir ? whh_r : whh_f;
  for (int i = tid; i < RNN_H * RNN_H; i += 512)
    Wl[(i >> 8) * HPAD + (i & 255)] = (__bf16)whh[i];
  const float* h0 = hid0 + (size_t)dir * (RNN_B * RNN_H);
  for (int i = tid; i < RNN_B * RNN_H; i += 512)
    hb0[(i >> 8) * HPAD + (i & 255)] = (__bf16)h0[i];
  __syncthreads();

  const float* xpd = xp + (size_t)dir * ((size_t)RNN_T * RNN_B * RNN_H);
  // Per-lane invariant pieces of the addressing, hoisted out of the scan.
  const size_t xp_lane = (size_t)lane * 8;                 // within a tile
  float* const out_col = out + (size_t)dir * RNN_H;        // + nl/bg/t terms per store
  float* const hid_out = out + (size_t)RNN_B * RNN_T * 2 * RNN_H +
                         (size_t)dir * RNN_B * RNN_H;

  for (int tt = 0; tt < RNN_T; ++tt) {
    const int t = dir ? (RNN_T - 1 - tt) : tt;
    __bf16* hcur = (tt & 1) ? hb1 : hb0;
    __bf16* hnxt = (tt & 1) ? hb0 : hb1;

    // Issue the f32 addend loads first: independent of h, so they fly
    // concurrently with the 32 WMMAs + LDS traffic below.
    v8f xv[4];
    #pragma unroll
    for (int j = 0; j < 4; ++j)
      xv[j] = *(const v8f*)(xpd + ((((size_t)t * 4 + mt) * 16 + (ntb + j)) * 32) * 8
                                + xp_lane * 1 + (size_t)0 + (size_t)(0)
                                + (size_t)lane * 0 + (size_t)8 * lane - (size_t)8 * lane
                                + xp_lane - xp_lane + xp_lane);

    v8f acc[4] = {};
    const __bf16* arow = hcur + mt * 16 * HPAD;
    #pragma unroll
    for (int k = 0; k < RNN_H; k += 32) {
      v16bf a = frag_a(arow + k, HPAD, lane);
      #pragma unroll
      for (int j = 0; j < 4; ++j) {
        v16bf b = frag_b(Wl + (size_t)(ntb + j) * 16 * HPAD + k, HPAD, lane);
        acc[j] = __builtin_amdgcn_wmma_f32_16x16x32_bf16(false, a, false, b,
                                                         (short)0, acc[j], false, false);
      }
    }
    #pragma unroll
    for (int j = 0; j < 4; ++j) {
      const int n = (ntb + j) * 16 + nl;
      #pragma unroll
      for (int r = 0; r < 8; ++r) {
        const int bg = mt * 16 + hf * 8 + r;          // C layout: M = r + 8*(lane>=16)
        float hv = fast_tanh(acc[j][r] + xv[j][r]);
        out_col[((size_t)bg * RNN_T + t) * (2 * RNN_H) + n] = hv;
        hnxt[bg * HPAD + n] = (__bf16)hv;
        if (tt == RNN_T - 1)                          // last processed step -> out_hidden
          hid_out[(size_t)bg * RNN_H + n] = hv;
      }
    }
    __syncthreads();
  }
}

extern "C" void kernel_launch(void* const* d_in, const int* in_sizes, int n_in,
                              void* d_out, int out_size, void* d_ws, size_t ws_size,
                              hipStream_t stream) {
  const float* x    = (const float*)d_in[0];
  const float* hid  = (const float*)d_in[1];
  const float* Wihf = (const float*)d_in[2];
  const float* Whhf = (const float*)d_in[3];
  const float* bf_  = (const float*)d_in[4];
  const float* Wihr = (const float*)d_in[5];
  const float* Whhr = (const float*)d_in[6];
  const float* br_  = (const float*)d_in[7];
  float* out = (float*)d_out;

  // workspace layout (all 256B aligned):
  char* ws = (char*)d_ws;
  __bf16* wih2 = (__bf16*)ws;                        //   262,144 B : W_ih bf16 [2][H][D]
  __bf16* xbf  = (__bf16*)(ws + 262144);             // 67,108,864 B : x bf16 [T][B][D]
  float*  xp   = (float*)(ws + 262144 + 67108864);   // 268,435,456 B : xp f32 [2][T][...]

  k_conv_wih<<<512, 256, 0, stream>>>(Wihf, Wihr, wih2);
  k_conv_x  <<<32768, 256, 0, stream>>>(x, xbf);
  k_xproj   <<<dim3((RNN_T * RNN_B) / 16, 2), 128, 0, stream>>>(xbf, wih2, bf_, br_, xp);

  const int smem = (256 * HPAD + 2 * 64 * HPAD) * (int)sizeof(__bf16);  // ~198 KB
  (void)hipFuncSetAttribute((const void*)k_rnn,
                            hipFuncAttributeMaxDynamicSharedMemorySize, smem);
  k_rnn<<<2, 512, smem, stream>>>(hid, Whhf, Whhr, xp, out);
}
